// TransformerBlock_8246337208415
// MI455X (gfx1250) — matrix operand order repaired and checked by reading, not compiled, not run on hardware
//
#include <hip/hip_runtime.h>
#include <hip/hip_bf16.h>

// ---------------------------------------------------------------------------
// Problem constants: B=32, S=577, D=768, H=12, d=64
// ---------------------------------------------------------------------------
constexpr int Bn   = 32;
constexpr int Sn   = 577;
constexpr int Dn   = 768;
constexpr int Hn   = 12;
constexpr int dh   = 64;
constexpr int MLPn = 3072;
constexpr int SPn  = 608;              // S padded to 19 * 32 (WMMA K blocks)
constexpr int TOKn = Bn * Sn;          // 18464 = 1154 * 16 exactly

typedef __bf16 bf16;
typedef __attribute__((ext_vector_type(16))) __bf16 v16bf;
typedef __attribute__((ext_vector_type(8)))  __bf16 bf16x8;
typedef __attribute__((ext_vector_type(8)))  float  v8f;

// ---------------------------------------------------------------------------
// WMMA wrapper: D = A(16x32 bf16) * B(32x16 bf16) + C(16x16 f32)
// ---------------------------------------------------------------------------
__device__ inline v8f wmma_bf16(v16bf a, v16bf b, v8f c) {
    return __builtin_amdgcn_wmma_f32_16x16x32_bf16(
        /*neg_a=*/false, a, /*neg_b=*/false, b,
        /*c_mod=*/(short)0, c, /*reuse_a=*/false, /*reuse_b=*/false);
}

__device__ inline v8f zero8f() {
    v8f z;
#pragma unroll
    for (int i = 0; i < 8; ++i) z[i] = 0.0f;
    return z;
}
__device__ inline v16bf zero16bf() {
    v16bf z;
#pragma unroll
    for (int i = 0; i < 16; ++i) z[i] = (__bf16)0.0f;
    return z;
}

// A-fragment (16x32 bf16, row-major src, leading dim ld):
// lane L: row = L%16; holds K = c..c+7 and c+16..c+23 with c = 8*(L/16).
__device__ inline v16bf load_a_frag(const bf16* p, int ld) {
    const int lane = threadIdx.x & 31;
    const int r = lane & 15;
    const int c = (lane >> 4) << 3;
    const bf16* q = p + r * ld + c;
    bf16x8 lo = *(const bf16x8*)(q);
    bf16x8 hi = *(const bf16x8*)(q + 16);
    v16bf a;
#pragma unroll
    for (int i = 0; i < 8; ++i) { a[i] = lo[i]; a[8 + i] = hi[i]; }
    return a;
}

// B-fragment (32x16 bf16) from a column-major / transposed source:
// element (k,n) lives at p[n*ld + k].
// lane L: col n = L%16; holds K = kh..kh+7 and kh+16..kh+23 with kh = 8*(L/16).
__device__ inline v16bf load_bt_frag(const bf16* p, int ld) {
    const int lane = threadIdx.x & 31;
    const int n  = lane & 15;
    const int kh = (lane >> 4) << 3;
    const bf16* q = p + n * ld + kh;
    bf16x8 lo = *(const bf16x8*)(q);
    bf16x8 hi = *(const bf16x8*)(q + 16);
    v16bf b;
#pragma unroll
    for (int i = 0; i < 8; ++i) { b[i] = lo[i]; b[8 + i] = hi[i]; }
    return b;
}

// ---------------------------------------------------------------------------
// Weight convert + transpose: out[g][n][k] = (bf16) in[g][k][n]
// ---------------------------------------------------------------------------
__global__ void wtrans_kernel(const float* __restrict__ in, bf16* __restrict__ out,
                              int batch, int K, int N) {
    long long total = (long long)batch * K * N;
    long long i = (long long)blockIdx.x * 256 + threadIdx.x;
    if (i >= total) return;
    int n = (int)(i % N);
    long long t = i / N;
    int k = (int)(t % K);
    int g = (int)(t / K);
    out[(long long)g * K * N + (long long)n * K + k] = (bf16)in[i];
}

// ---------------------------------------------------------------------------
// LayerNorm: one 256-thread block per token row of 768; writes bf16.
// ---------------------------------------------------------------------------
__global__ __launch_bounds__(256) void ln_kernel(const float* __restrict__ x,
                                                 const float* __restrict__ g,
                                                 const float* __restrict__ b,
                                                 bf16* __restrict__ out) {
    const int row = blockIdx.x;
    const int t = threadIdx.x;
    const float* xr = x + (long long)row * Dn;
    float v0 = xr[t], v1 = xr[t + 256], v2 = xr[t + 512];
    __shared__ float red[256];
    red[t] = v0 + v1 + v2;
    __syncthreads();
#pragma unroll
    for (int off = 128; off > 0; off >>= 1) {
        if (t < off) red[t] += red[t + off];
        __syncthreads();
    }
    const float mu = red[0] * (1.0f / Dn);
    __syncthreads();
    float d0 = v0 - mu, d1 = v1 - mu, d2 = v2 - mu;
    red[t] = d0 * d0 + d1 * d1 + d2 * d2;
    __syncthreads();
#pragma unroll
    for (int off = 128; off > 0; off >>= 1) {
        if (t < off) red[t] += red[t + off];
        __syncthreads();
    }
    const float rstd = rsqrtf(red[0] * (1.0f / Dn) + 1e-5f);
    bf16* orow = out + (long long)row * Dn;
    orow[t]       = (bf16)(d0 * rstd * g[t]       + b[t]);
    orow[t + 256] = (bf16)(d1 * rstd * g[t + 256] + b[t + 256]);
    orow[t + 512] = (bf16)(d2 * rstd * g[t + 512] + b[t + 512]);
}

// ---------------------------------------------------------------------------
// QKV projection: grid (SP/16, B*H), block 128 (4 waves, one 16-col tile each)
// q,k stored [bh][SP][64] bf16; v stored transposed [bh][64][SP] bf16.
// ---------------------------------------------------------------------------
__global__ __launch_bounds__(128) void qkv_kernel(const bf16* __restrict__ xh,
                                                  const bf16* __restrict__ wqt, const float* __restrict__ bq,
                                                  const bf16* __restrict__ wkt, const float* __restrict__ bk,
                                                  const bf16* __restrict__ wvt, const float* __restrict__ bv,
                                                  bf16* __restrict__ qo, bf16* __restrict__ ko,
                                                  bf16* __restrict__ vto) {
    const int stile = blockIdx.x;
    const int bh = blockIdx.y;
    const int bb = bh / Hn, h = bh % Hn;
    const int wv  = threadIdx.x >> 5;
    const int lane = threadIdx.x & 31;
    const int r = lane & 15, hi = lane >> 4, col = lane & 15;
    const int row0 = stile * 16;
    const int n0 = wv * 16;

    v16bf a0 = zero16bf(), a1 = zero16bf();
    if (row0 + r < Sn) {
        const bf16* abase = xh + ((long long)bb * Sn + row0) * Dn + h * dh;
        a0 = load_a_frag(abase, Dn);
        a1 = load_a_frag(abase + 32, Dn);
    }

    const bf16* wqp = wqt + h * dh * dh + n0 * dh;
    const bf16* wkp = wkt + h * dh * dh + n0 * dh;
    const bf16* wvp = wvt + h * dh * dh + n0 * dh;

    v8f aq = zero8f(), ak = zero8f(), av = zero8f();
    aq = wmma_bf16(a0, load_bt_frag(wqp, dh), aq);
    aq = wmma_bf16(a1, load_bt_frag(wqp + 32, dh), aq);
    ak = wmma_bf16(a0, load_bt_frag(wkp, dh), ak);
    ak = wmma_bf16(a1, load_bt_frag(wkp + 32, dh), ak);
    av = wmma_bf16(a0, load_bt_frag(wvp, dh), av);
    av = wmma_bf16(a1, load_bt_frag(wvp + 32, dh), av);

    const float biasq = bq[h * dh + n0 + col];
    const float biask = bk[h * dh + n0 + col];
    const float biasv = bv[h * dh + n0 + col];
#pragma unroll
    for (int j = 0; j < 8; ++j) {
        const int rowg = row0 + j + hi * 8;
        const long long base = ((long long)bh * SPn + rowg) * dh + n0 + col;
        qo[base] = (bf16)(aq[j] + biasq);
        ko[base] = (bf16)(ak[j] + biask);
        vto[((long long)bh * dh + n0 + col) * SPn + rowg] = (bf16)(av[j] + biasv);
    }
}

// ---------------------------------------------------------------------------
// Flash attention: grid (SP/16, B*H), 1 wave per block.
// Online softmax over 19 key blocks of 32; writes res1 = x + attn_out (f32).
// ---------------------------------------------------------------------------
__global__ __launch_bounds__(32) void attn_kernel(const bf16* __restrict__ qb,
                                                  const bf16* __restrict__ kb,
                                                  const bf16* __restrict__ vtb,
                                                  const float* __restrict__ x,
                                                  float* __restrict__ res1) {
    const int qt = blockIdx.x;
    const int bh = blockIdx.y;
    const int bb = bh / Hn, h = bh % Hn;
    const int lane = threadIdx.x;
    const int hi = lane >> 4, col = lane & 15;

    const bf16* qp = qb + ((long long)bh * SPn + qt * 16) * dh;
    v16bf qa0 = load_a_frag(qp, dh);
    v16bf qa1 = load_a_frag(qp + 32, dh);

    const bf16* kbase  = kb  + (long long)bh * SPn * dh;
    const bf16* vtbase = vtb + (long long)bh * dh * SPn;

    float m[8], l[8];
    v8f o[4];
#pragma unroll
    for (int j = 0; j < 8; ++j) { m[j] = -1e30f; l[j] = 0.0f; }
#pragma unroll
    for (int t = 0; t < 4; ++t) o[t] = zero8f();

    __shared__ __align__(16) bf16 pl[16][32];

    for (int blk = 0; blk < SPn / 32; ++blk) {
        const int K0 = blk * 32;
        const bf16* kp = kbase + (long long)K0 * dh;

        // speculative prefetch of the next key block (global_prefetch_b8)
        if (blk + 1 < SPn / 32)
            __builtin_prefetch(kbase + (long long)(K0 + 32 + lane) * dh, 0, 1);

        v8f s0 = zero8f(), s1 = zero8f();
        s0 = wmma_bf16(qa0, load_bt_frag(kp, dh), s0);
        s0 = wmma_bf16(qa1, load_bt_frag(kp + 32, dh), s0);
        s1 = wmma_bf16(qa0, load_bt_frag(kp + 16 * dh, dh), s1);
        s1 = wmma_bf16(qa1, load_bt_frag(kp + 16 * dh + 32, dh), s1);

        const bool v0ok = (K0 + col) < Sn;
        const bool v1ok = (K0 + 16 + col) < Sn;
#pragma unroll
        for (int j = 0; j < 8; ++j) {
            float t0 = v0ok ? s0[j] * 0.125f : -1e30f;   // 1/sqrt(64)
            float t1 = v1ok ? s1[j] * 0.125f : -1e30f;
            float mx = fmaxf(t0, t1);
#pragma unroll
            for (int msk = 1; msk < 16; msk <<= 1)
                mx = fmaxf(mx, __shfl_xor(mx, msk, 32));
            const float mn = fmaxf(m[j], mx);
            const float alpha = __expf(m[j] - mn);
            const float e0 = __expf(t0 - mn);
            const float e1 = __expf(t1 - mn);
            float rs = e0 + e1;
#pragma unroll
            for (int msk = 1; msk < 16; msk <<= 1)
                rs += __shfl_xor(rs, msk, 32);
            l[j] = l[j] * alpha + rs;
            m[j] = mn;
            pl[j + hi * 8][col]      = (bf16)e0;
            pl[j + hi * 8][16 + col] = (bf16)e1;
#pragma unroll
            for (int t = 0; t < 4; ++t) o[t][j] = o[t][j] * alpha;
        }

        // Re-swizzle P (C-layout in LDS) into an A-fragment; same-wave DS ops
        // are in-order so no barrier is needed.
        v16bf pa = load_a_frag(&pl[0][0], 32);
#pragma unroll
        for (int t = 0; t < 4; ++t)
            o[t] = wmma_bf16(pa, load_bt_frag(vtbase + (long long)(t * 16) * SPn + K0, SPn), o[t]);
    }

#pragma unroll
    for (int j = 0; j < 8; ++j) l[j] = 1.0f / l[j];
#pragma unroll
    for (int t = 0; t < 4; ++t) {
#pragma unroll
        for (int j = 0; j < 8; ++j) {
            const int rq = qt * 16 + j + hi * 8;
            if (rq < Sn) {
                const long long idx = ((long long)bb * Sn + rq) * Dn + h * dh + t * 16 + col;
                res1[idx] = x[idx] + o[t][j] * l[j];
            }
        }
    }
}

// ---------------------------------------------------------------------------
// FC1: [18464,768] x [768,3072] + b1, exact GELU -> bf16 mid.
// grid (1154, 3), block 256 (8 waves, each a 16x128 tile).
// A tile (16x768 bf16 = 24 KB) staged once in LDS, shared by all 8 waves.
// ---------------------------------------------------------------------------
__global__ __launch_bounds__(256) void fc1_kernel(const bf16* __restrict__ hb,
                                                  const bf16* __restrict__ w1t,
                                                  const float* __restrict__ b1,
                                                  bf16* __restrict__ mid) {
    extern __shared__ __align__(16) bf16 atile1[];           // 16*Dn bf16
    const int mt = blockIdx.x;
    const int wv = threadIdx.x >> 5;
    const int lane = threadIdx.x & 31;
    const int hi = lane >> 4, col = lane & 15;
    const int n0 = blockIdx.y * 1024 + wv * 128;

    // cooperative stage of the shared A tile (b128 copies)
    {
        const bf16x8* src = (const bf16x8*)(hb + (long long)mt * 16 * Dn);
        bf16x8* dst = (bf16x8*)atile1;
#pragma unroll
        for (int i = threadIdx.x; i < 16 * Dn / 8; i += 256) dst[i] = src[i];
    }
    __syncthreads();

    v8f acc[8];
#pragma unroll
    for (int t = 0; t < 8; ++t) acc[t] = zero8f();

    for (int ks = 0; ks < Dn / 32; ++ks) {
        v16bf a = load_a_frag(atile1 + ks * 32, Dn);
        // prefetch the weight stream ~512B ahead (global_prefetch_b8)
        if (ks + 8 < Dn / 32)
            __builtin_prefetch(w1t + (long long)(n0 + col) * Dn + (ks + 8) * 32, 0, 1);
#pragma unroll
        for (int t = 0; t < 8; ++t)
            acc[t] = wmma_bf16(a, load_bt_frag(w1t + (long long)(n0 + t * 16) * Dn + ks * 32, Dn), acc[t]);
    }
#pragma unroll
    for (int t = 0; t < 8; ++t) {
        const float bias = b1[n0 + t * 16 + col];
#pragma unroll
        for (int j = 0; j < 8; ++j) {
            const int r = mt * 16 + j + hi * 8;
            float u = acc[t][j] + bias;
            float ge = 0.5f * u * (1.0f + erff(u * 0.70710678118f));
            mid[(long long)r * MLPn + n0 + t * 16 + col] = (bf16)ge;
        }
    }
}

// ---------------------------------------------------------------------------
// FC2: [18464,3072] x [3072,768] + b2 + res1 -> d_out f32.
// grid (1154, 1), block 192 (6 waves, each a 16x128 tile).
// A tile (16x3072 bf16 = 96 KB) staged once in LDS (WGP has 320 KB).
// ---------------------------------------------------------------------------
__global__ __launch_bounds__(192) void fc2_kernel(const bf16* __restrict__ mid,
                                                  const bf16* __restrict__ w2t,
                                                  const float* __restrict__ b2,
                                                  const float* __restrict__ res1,
                                                  float* __restrict__ out) {
    extern __shared__ __align__(16) bf16 atile2[];           // 16*MLPn bf16
    const int mt = blockIdx.x;
    const int wv = threadIdx.x >> 5;
    const int lane = threadIdx.x & 31;
    const int hi = lane >> 4, col = lane & 15;
    const int n0 = wv * 128;

    {
        const bf16x8* src = (const bf16x8*)(mid + (long long)mt * 16 * MLPn);
        bf16x8* dst = (bf16x8*)atile2;
        for (int i = threadIdx.x; i < 16 * MLPn / 8; i += 192) dst[i] = src[i];
    }
    __syncthreads();

    v8f acc[8];
#pragma unroll
    for (int t = 0; t < 8; ++t) acc[t] = zero8f();

    for (int ks = 0; ks < MLPn / 32; ++ks) {
        v16bf a = load_a_frag(atile2 + ks * 32, MLPn);
        if (ks + 8 < MLPn / 32)
            __builtin_prefetch(w2t + (long long)(n0 + col) * MLPn + (ks + 8) * 32, 0, 1);
#pragma unroll
        for (int t = 0; t < 8; ++t)
            acc[t] = wmma_bf16(a, load_bt_frag(w2t + (long long)(n0 + t * 16) * MLPn + ks * 32, MLPn), acc[t]);
    }
#pragma unroll
    for (int t = 0; t < 8; ++t) {
        const float bias = b2[n0 + t * 16 + col];
#pragma unroll
        for (int j = 0; j < 8; ++j) {
            const int r = mt * 16 + j + hi * 8;
            const long long idx = (long long)r * Dn + n0 + t * 16 + col;
            out[idx] = acc[t][j] + bias + res1[idx];
        }
    }
}

// ---------------------------------------------------------------------------
// Launch orchestration
// ---------------------------------------------------------------------------
extern "C" void kernel_launch(void* const* d_in, const int* in_sizes, int n_in,
                              void* d_out, int out_size, void* d_ws, size_t ws_size,
                              hipStream_t stream) {
    const float* x     = (const float*)d_in[0];
    const float* ln1_g = (const float*)d_in[1];
    const float* ln1_b = (const float*)d_in[2];
    const float* Wq    = (const float*)d_in[3];
    const float* bq    = (const float*)d_in[4];
    const float* Wk    = (const float*)d_in[5];
    const float* bk    = (const float*)d_in[6];
    const float* Wv    = (const float*)d_in[7];
    const float* bv    = (const float*)d_in[8];
    const float* ln2_g = (const float*)d_in[9];
    const float* ln2_b = (const float*)d_in[10];
    const float* W1    = (const float*)d_in[11];
    const float* b1    = (const float*)d_in[12];
    const float* W2    = (const float*)d_in[13];
    const float* b2    = (const float*)d_in[14];
    float* out = (float*)d_out;

    // Workspace layout (bytes):
    //   res1 f32 [18464,768]                      : 56,721,408
    //   xh/h bf16 [18464,768] (reused for LN1/LN2): 28,360,704
    //   q,k,vT bf16 (3 x [384,608,64]) overlapped
    //     with mid bf16 [18464,3072]              : 113,442,816
    //   w1t, w2t, wq/k/v-T bf16                   : ~9.7 MB
    char* ws = (char*)d_ws;
    const size_t off_xh  = (size_t)TOKn * Dn * 4;
    const size_t off_qkv = off_xh + (size_t)TOKn * Dn * 2;
    const size_t off_w   = off_qkv + (size_t)TOKn * MLPn * 2;
    const size_t qkv_elems = (size_t)Bn * Hn * SPn * dh;

    float* res1 = (float*)(ws);
    bf16*  xh   = (bf16*)(ws + off_xh);
    bf16*  qb   = (bf16*)(ws + off_qkv);
    bf16*  kbuf = qb + qkv_elems;
    bf16*  vtb  = kbuf + qkv_elems;
    bf16*  mid  = (bf16*)(ws + off_qkv);          // overlaps q/k/v (dead by then)
    bf16*  w1t  = (bf16*)(ws + off_w);
    bf16*  w2t  = w1t + (size_t)Dn * MLPn;
    bf16*  wqt  = w2t + (size_t)MLPn * Dn;
    bf16*  wkt  = wqt + (size_t)Hn * dh * dh;
    bf16*  wvt  = wkt + (size_t)Hn * dh * dh;

    // 0) weight convert+transpose (f32 -> bf16, [k][n] -> [n][k])
    {
        int nW1 = Dn * MLPn;                       // 2,359,296
        wtrans_kernel<<<(nW1 + 255) / 256, 256, 0, stream>>>(W1, w1t, 1, Dn, MLPn);
        wtrans_kernel<<<(nW1 + 255) / 256, 256, 0, stream>>>(W2, w2t, 1, MLPn, Dn);
        int nWh = Hn * dh * dh;                    // 49,152
        wtrans_kernel<<<(nWh + 255) / 256, 256, 0, stream>>>(Wq, wqt, Hn, dh, dh);
        wtrans_kernel<<<(nWh + 255) / 256, 256, 0, stream>>>(Wk, wkt, Hn, dh, dh);
        wtrans_kernel<<<(nWh + 255) / 256, 256, 0, stream>>>(Wv, wvt, Hn, dh, dh);
    }

    // 1) LN1
    ln_kernel<<<TOKn, 256, 0, stream>>>(x, ln1_g, ln1_b, xh);

    // 2) QKV projections
    qkv_kernel<<<dim3(SPn / 16, Bn * Hn), 128, 0, stream>>>(xh, wqt, bq, wkt, bk, wvt, bv,
                                                            qb, kbuf, vtb);

    // 3) Flash attention + residual -> res1
    attn_kernel<<<dim3(SPn / 16, Bn * Hn), 32, 0, stream>>>(qb, kbuf, vtb, x, res1);

    // 4) LN2
    ln_kernel<<<TOKn, 256, 0, stream>>>(res1, ln2_g, ln2_b, xh);

    // 5) FC1 + GELU (24 KB dynamic LDS for the shared A tile)
    fc1_kernel<<<dim3(TOKn / 16, MLPn / 1024), 256, 16 * Dn * sizeof(bf16), stream>>>(xh, w1t, b1, mid);

    // 6) FC2 + bias + residual -> out (96 KB dynamic LDS for the shared A tile)
    fc2_kernel<<<dim3(TOKn / 16, 1), 192, 16 * MLPn * sizeof(bf16), stream>>>(mid, w2t, b2, res1, out);
}
